// SpikingTemporalEncoder_39642548142761
// MI455X (gfx1250) — matrix operand order repaired and checked by reading, not compiled, not run on hardware
//
#include <hip/hip_runtime.h>
#include <hip/hip_bf16.h>
#include <math.h>

typedef __attribute__((ext_vector_type(16))) __bf16 v16bf;
typedef __attribute__((ext_vector_type(8)))  float  v8f;

#define B_     64
#define T_     2048
#define IN_    4
#define H0_    64
#define D_     128
#define HD_    256
#define DEPTH_ 4
#define M_     (B_*T_)

// ---------------------------------------------------------------- helpers

__device__ __forceinline__ unsigned short f32_to_bf16_rne(float f) {
  unsigned int u = __float_as_uint(f);
  u += 0x7fffu + ((u >> 16) & 1u);   // round-to-nearest-even
  return (unsigned short)(u >> 16);
}

union FragBF { v16bf v; uint4 q[2]; };

__device__ __forceinline__ v8f v8f_zero() {
  v8f z = {0.f,0.f,0.f,0.f,0.f,0.f,0.f,0.f};
  return z;
}

// ------------------------------------------------- weight convert+transpose
// W: [depth, K, N] fp32  ->  WT: [depth, N, K] bf16
__global__ void convert_transpose_kernel(const float* __restrict__ W,
                                         unsigned short* __restrict__ WT,
                                         int depth, int K, int N) {
  int i = blockIdx.x * blockDim.x + threadIdx.x;
  int total = depth * K * N;
  if (i >= total) return;
  int d = i / (K * N);
  int r = i - d * (K * N);
  int k = r / N;
  int n = r - k * N;
  WT[(size_t)d * K * N + (size_t)n * K + k] = f32_to_bf16_rne(W[i]);
}

// ---------------------------------------------------------------- projector
// x = gelu_exact(hist @ pw1 + pb1) @ pw2 + pb2 ; one block per (b,t) row
__global__ __launch_bounds__(128)
void projector_kernel(const float* __restrict__ hist,   // [M,4]
                      const float* __restrict__ pw1,    // [4,64]
                      const float* __restrict__ pb1,    // [64]
                      const float* __restrict__ pw2,    // [64,128]
                      const float* __restrict__ pb2,    // [128]
                      float* __restrict__ X,            // [M,128] fp32
                      unsigned short* __restrict__ Xbf) // [M,128] bf16
{
  __shared__ float h0[H0_];
  __shared__ float xin[IN_];
  const int row = blockIdx.x;
  const int tid = threadIdx.x;
  if (tid < IN_) xin[tid] = hist[(size_t)row * IN_ + tid];
  __syncthreads();
  if (tid < H0_) {
    float a = pb1[tid];
#pragma unroll
    for (int k = 0; k < IN_; ++k) a += xin[k] * pw1[k * H0_ + tid];
    h0[tid] = 0.5f * a * (1.0f + erff(a * 0.70710678118654752440f));
  }
  __syncthreads();
  float a = pb2[tid];
#pragma unroll 8
  for (int k = 0; k < H0_; ++k) a += h0[k] * pw2[k * D_ + tid];
  X[(size_t)row * D_ + tid]   = a;
  Xbf[(size_t)row * D_ + tid] = f32_to_bf16_rne(a);
}

// ------------------------------------------------------- fused GEMM+bias+LN
// Y[M,N] = LayerNorm_rowwise(Abf[M,K] @ W[K,N] + bias) * gamma + beta
// WT is [N,K] bf16 (transposed).  One workgroup = WAVES waves, MT = WAVES*16
// rows; each wave owns 16 full rows across all N (LN reduction stays in-wave).
// A tile is staged with GLOBAL_LOAD_ASYNC_TO_LDS_B128 (ASYNCcnt-tracked DMA,
// no VGPR round trip), completed with s_wait_asynccnt before the barrier.
template<int K, int N, int WAVES>
__global__ __launch_bounds__(WAVES*32)
void gemm_bias_ln_kernel(const unsigned short* __restrict__ Abf,
                         const unsigned short* __restrict__ WT,
                         const float* __restrict__ bias,
                         const float* __restrict__ gamma,
                         const float* __restrict__ beta,
                         float* __restrict__ Y, int M)
{
  constexpr int THREADS = WAVES * 32;
  constexpr int MT = WAVES * 16;
  constexpr int AP = K + 8;          // padded LDS row (ushorts) -> no bank clash
  constexpr int NT = N / 16;
  extern __shared__ __align__(16) unsigned short sA[];   // [MT][AP] bf16

  const int tid   = threadIdx.x;
  const int lane  = tid & 31;
  const int wv    = tid >> 5;
  const int l15   = lane & 15;
  const int khalf = lane >> 4;
  const int m0    = blockIdx.x * MT;
  (void)M;

  __builtin_prefetch(WT, 0, 0);      // global_prefetch of L2-resident weights

  // stage A tile (bf16) into LDS via async DMA, 16B per lane per op
#pragma unroll
  for (int i = tid; i < MT * (K / 8); i += THREADS) {
    const int r = i / (K / 8);
    const int c = (i % (K / 8)) * 8;
    const unsigned short* gp = Abf + (size_t)(m0 + r) * K + c;
    unsigned lds_off = (unsigned)(uintptr_t)(sA + r * AP + c);
    asm volatile("global_load_async_to_lds_b128 %0, %1, off"
                 :: "v"(lds_off), "v"(gp) : "memory");
  }
  asm volatile("s_wait_asynccnt 0x0" ::: "memory");
  __syncthreads();

  v8f acc[NT];
#pragma unroll
  for (int nt = 0; nt < NT; ++nt) acc[nt] = v8f_zero();

  const int arow = wv * 16 + l15;    // A-matrix row for this lane
#pragma unroll
  for (int kc = 0; kc < K / 32; ++kc) {
    // A frag, contiguous k order (matches B): lane m=l15, k = kc*32 + khalf*16 + {0..15}
    FragBF a;
    const int ka = kc * 32 + khalf * 16;
    a.q[0] = *(const uint4*)(sA + arow * AP + ka);
    a.q[1] = *(const uint4*)(sA + arow * AP + ka + 8);
    const int kb = kc * 32 + khalf * 16;   // B frag: n=l15, k = kb + e (contig)
#pragma unroll
    for (int nt = 0; nt < NT; ++nt) {
      FragBF b;
      const unsigned short* wp = WT + (size_t)(nt * 16 + l15) * K + kb;
      b.q[0] = *(const uint4*)(wp);
      b.q[1] = *(const uint4*)(wp + 8);
      acc[nt] = __builtin_amdgcn_wmma_f32_16x16x32_bf16(
          false, a.v, false, b.v, (short)0, acc[nt], false, false);
    }
  }

  float bcol[NT], gcol[NT], btcol[NT];
#pragma unroll
  for (int nt = 0; nt < NT; ++nt) {
    const int n = nt * 16 + l15;
    bcol[nt] = bias[n]; gcol[nt] = gamma[n]; btcol[nt] = beta[n];
  }

  const float invN = 1.0f / (float)N;
  // C/D layout: vgpr r holds row (r + 8*khalf), col = l15 of each 16-wide tile
#pragma unroll
  for (int r = 0; r < 8; ++r) {
    float s1 = 0.f, s2 = 0.f;
#pragma unroll
    for (int nt = 0; nt < NT; ++nt) {
      const float v = acc[nt][r] + bcol[nt];
      s1 += v; s2 += v * v;
    }
#pragma unroll
    for (int msk = 1; msk < 16; msk <<= 1) {   // butterfly within 16-lane half
      s1 += __shfl_xor(s1, msk, 32);
      s2 += __shfl_xor(s2, msk, 32);
    }
    const float mean = s1 * invN;
    const float var  = s2 * invN - mean * mean;
    const float rstd = rsqrtf(var + 1e-5f);
    const int grow = m0 + wv * 16 + khalf * 8 + r;
    float* yrow = Y + (size_t)grow * N;
#pragma unroll
    for (int nt = 0; nt < NT; ++nt) {
      const float v = acc[nt][r] + bcol[nt];
      yrow[nt * 16 + l15] = (v - mean) * rstd * gcol[nt] + btcol[nt];
    }
  }
}

// ----------------------------------------------------------------- LIF scan
// v' = v + (x - v)/2 ; s = (v' >= 1) ; v = s ? 0 : v'   (hard reset, forward)
// One thread per channel (b,d), sequential over T; coalesced across d.
// Unrolled x16 so the load clause keeps ~16 independent L2 hits in flight;
// explicit prefetch 64 steps ahead (doesn't consume LOADcnt).
template<bool FUSE_RES>
__global__ void lif_kernel(const float* __restrict__ Yin,   // [B,T,Dm]
                           unsigned short* __restrict__ Sout,// spikes bf16
                           const float* Resid,               // [B,T,Dm]
                           float* Xout,                      // fp32 x (fused)
                           unsigned short* __restrict__ Xbf, // bf16 x (fused)
                           int T, int Dm, int C)
{
  const int c = blockIdx.x * blockDim.x + threadIdx.x;
  if (c >= C) return;
  const int b = c / Dm;
  const int d = c - b * Dm;
  size_t idx = (size_t)b * T * Dm + d;
  const size_t pfj = (size_t)64 * Dm;
  float v = 0.f;
#pragma unroll 16
  for (int t = 0; t < T; ++t, idx += Dm) {
    if (t + 64 < T) __builtin_prefetch(Yin + idx + pfj, 0, 0);
    const float x = Yin[idx];
    v = 0.5f * (v + x);
    const bool fire = (v >= 1.0f);
    if (FUSE_RES) {
      const float o = Resid[idx] + (fire ? 1.0f : 0.0f);
      Xout[idx] = o;
      Xbf[idx]  = f32_to_bf16_rne(o);
    } else {
      Sout[idx] = fire ? (unsigned short)0x3F80 : (unsigned short)0;
    }
    v = fire ? 0.0f : v;
  }
}

// ------------------------------------------------------------- last pooling
__global__ void last_pool_kernel(const float* __restrict__ X,
                                 float* __restrict__ out, int n) {
  const int i = blockIdx.x * blockDim.x + threadIdx.x;
  if (i >= n) return;
  const int b = i / D_;
  const int d = i - b * D_;
  out[i] = X[((size_t)b * T_ + (T_ - 1)) * D_ + d];
}

// -------------------------------------------------------------------- launch
extern "C" void kernel_launch(void* const* d_in, const int* in_sizes, int n_in,
                              void* d_out, int out_size, void* d_ws, size_t ws_size,
                              hipStream_t stream)
{
  (void)in_sizes; (void)n_in; (void)out_size; (void)ws_size;
  const float* hist  = (const float*)d_in[0];
  const float* pw1   = (const float*)d_in[1];
  const float* pb1   = (const float*)d_in[2];
  const float* pw2   = (const float*)d_in[3];
  const float* pb2   = (const float*)d_in[4];
  const float* fc1_w = (const float*)d_in[5];
  const float* fc1_b = (const float*)d_in[6];
  const float* n1_g  = (const float*)d_in[7];
  const float* n1_b  = (const float*)d_in[8];
  const float* fc2_w = (const float*)d_in[9];
  const float* fc2_b = (const float*)d_in[10];
  const float* n2_g  = (const float*)d_in[11];
  const float* n2_b  = (const float*)d_in[12];

  char* ws = (char*)d_ws;
  float*          xbuf = (float*)ws;          ws += (size_t)M_ * D_  * sizeof(float);
  unsigned short* xbf  = (unsigned short*)ws; ws += (size_t)M_ * D_  * 2;
  float*          ybuf = (float*)ws;          ws += (size_t)M_ * HD_ * sizeof(float);
  unsigned short* sbuf = (unsigned short*)ws; ws += (size_t)M_ * HD_ * 2;
  unsigned short* w1T  = (unsigned short*)ws; ws += (size_t)DEPTH_ * D_ * HD_ * 2;
  unsigned short* w2T  = (unsigned short*)ws; ws += (size_t)DEPTH_ * HD_ * D_ * 2;

  // 1) weights -> bf16, transposed to [N][K]
  {
    const int tot = DEPTH_ * D_ * HD_;
    convert_transpose_kernel<<<(tot + 255) / 256, 256, 0, stream>>>(fc1_w, w1T, DEPTH_, D_, HD_);
    convert_transpose_kernel<<<(tot + 255) / 256, 256, 0, stream>>>(fc2_w, w2T, DEPTH_, HD_, D_);
  }

  // 2) input projector -> x (fp32 + bf16 mirror)
  projector_kernel<<<M_, 128, 0, stream>>>(hist, pw1, pb1, pw2, pb2, xbuf, xbf);

  // 3) DEPTH spiking blocks
  const size_t shm1 = (size_t)(8 * 16) * (128 + 8) * 2;  // 34816 B
  const size_t shm2 = (size_t)(4 * 16) * (256 + 8) * 2;  // 33792 B
  for (int i = 0; i < DEPTH_; ++i) {
    gemm_bias_ln_kernel<128, 256, 8><<<M_ / 128, 256, shm1, stream>>>(
        xbf, w1T + (size_t)i * D_ * HD_,
        fc1_b + i * HD_, n1_g + i * HD_, n1_b + i * HD_, ybuf, M_);
    lif_kernel<false><<<(B_ * HD_ + 255) / 256, 256, 0, stream>>>(
        ybuf, sbuf, nullptr, nullptr, nullptr, T_, HD_, B_ * HD_);
    gemm_bias_ln_kernel<256, 128, 4><<<M_ / 64, 128, shm2, stream>>>(
        sbuf, w2T + (size_t)i * HD_ * D_,
        fc2_b + i * D_, n2_g + i * D_, n2_b + i * D_, ybuf, M_);
    lif_kernel<true><<<(B_ * D_ + 255) / 256, 256, 0, stream>>>(
        ybuf, nullptr, xbuf, xbuf, xbf, T_, D_, B_ * D_);
  }

  // 4) last-timestep pooling -> out [B, D]
  last_pool_kernel<<<(B_ * D_ + 255) / 256, 256, 0, stream>>>(xbuf, (float*)d_out, B_ * D_);
}
